// KrigingLocalityAdapter_18966575579414
// MI455X (gfx1250) — matrix operand order repaired and checked by reading, not compiled, not run on hardware
//
#include <hip/hip_runtime.h>
#include <hip/hip_bf16.h>
#include <math.h>

// Problem constants (from reference): B=64, T=4096, D=256, NS=27, NT=8, K=64, NB=256
#define B_    64
#define T_    4096
#define D_    256
#define NS_   27
#define NT_   8
#define NB_   256

#define KEFF  288   // effective K for GEMM-1: 256 hidden + 8 time + 24 zero pad (9 k-tiles of 32)
#define N1    512   // hidden width of corr MLP
#define N2    256   // output width
#define MT    128   // M tile (tokens) per workgroup
#define WS_HALF 16384  // bf16 elems per weight-slice buffer (512 rows x 32 k)

typedef __attribute__((ext_vector_type(16))) __bf16 v16bf;
typedef __attribute__((ext_vector_type(8)))  float  v8f;

union V16B { int4 q[2]; v16bf v; };   // 32 bytes: one wave32 WMMA A/B fragment per lane

__device__ __forceinline__ float gelu_exact(float x) {
    return 0.5f * x * (1.0f + erff(x * 0.70710678118654752440f));
}

// CDNA5 async L2->LDS copy (ASYNCcnt path, ISA cdna5 ch.10 / 15.18.3).
// Each enabled lane copies 16B from its global address to its LDS address.
// VDST = per-lane LDS byte address (low 32 bits of generic pointer), GV mode.
__device__ __forceinline__ void async_copy_b128(void* lds_dst, const void* gsrc) {
    asm volatile("global_load_async_to_lds_b128 %0, %1, off"
                 :: "v"((unsigned int)(unsigned long long)(uintptr_t)lds_dst),
                    "v"((unsigned long long)(uintptr_t)gsrc)
                 : "memory");
}
__device__ __forceinline__ void wait_async0() {
#if __has_builtin(__builtin_amdgcn_s_wait_asynccnt)
    __builtin_amdgcn_s_wait_asynccnt(0);
#else
    asm volatile("s_wait_asynccnt 0x0" ::: "memory");
#endif
}

// ---------------------------------------------------------------------------
// Kernel 1: tiny path — phi(static), phi(bank), v, softmax attention, alpha[b],
// plus static_proj[b][d]. All O(MFLOP); one workgroup.
// ---------------------------------------------------------------------------
__global__ __launch_bounds__(256) void small_path_kernel(
    const float* __restrict__ sf,      const float* __restrict__ bank,
    const float* __restrict__ phi_w1,  const float* __restrict__ phi_b1,
    const float* __restrict__ phi_w2,  const float* __restrict__ phi_b2,
    const float* __restrict__ vproj_w, const float* __restrict__ vproj_b,
    const float* __restrict__ rho_w1,  const float* __restrict__ rho_b1,
    const float* __restrict__ rho_w2,  const float* __restrict__ rho_b2,
    const float* __restrict__ stat_w,  const float* __restrict__ stat_b,
    float* __restrict__ sp_ws, float* __restrict__ alpha_ws)
{
    extern __shared__ char smem[];
    float* qs = (float*)smem;          // 64 x 64
    float* ks = qs + 64 * 64;          // 256 x 64
    float* vs = ks + 256 * 64;         // 256 x 64
    const int tid = threadIdx.x;

    // phi over [static_features ; bank]  (320 rows, NS=27 in, 64 out)
    for (int i = tid; i < 64 + NB_; i += 256) {
        const float* src = (i < 64) ? (sf + i * NS_) : (bank + (i - 64) * NS_);
        float hh[128];
        for (int j = 0; j < 128; ++j) {
            float a = phi_b1[j];
            for (int s = 0; s < NS_; ++s) a += phi_w1[j * NS_ + s] * src[s];
            hh[j] = gelu_exact(a);
        }
        for (int j2 = 0; j2 < 64; ++j2) {
            float o = phi_b2[j2];
            for (int j = 0; j < 128; ++j) o += phi_w2[j2 * 128 + j] * hh[j];
            if (i < 64) qs[i * 64 + j2] = o; else ks[(i - 64) * 64 + j2] = o;
        }
        if (i >= 64) {
            for (int j2 = 0; j2 < 64; ++j2) {
                float o = vproj_b[j2];
                for (int s = 0; s < NS_; ++s) o += vproj_w[j2 * NS_ + s] * src[s];
                vs[(i - 64) * 64 + j2] = o;
            }
        }
    }

    // static_proj[b][d] -> workspace (consumed by prep_weights_kernel)
    for (int i = tid; i < B_ * D_; i += 256) {
        int b = i >> 8, d = i & 255;
        float a = stat_b[d];
        for (int s = 0; s < NS_; ++s) a += stat_w[d * NS_ + s] * sf[b * NS_ + s];
        sp_ws[i] = a;
    }
    __syncthreads();

    // attention + gate: one thread per batch row
    if (tid < B_) {
        float sc[NB_];
        float mx = -1e30f;
        for (int j = 0; j < NB_; ++j) {
            float s = 0.f;
            for (int c = 0; c < 64; ++c) s += qs[tid * 64 + c] * ks[j * 64 + c];
            s *= 0.125f;                       // 1/sqrt(K=64)
            sc[j] = s; mx = fmaxf(mx, s);
        }
        float sum = 0.f;
        for (int j = 0; j < NB_; ++j) { float e = expf(sc[j] - mx); sc[j] = e; sum += e; }
        float inv = 1.f / sum;
        float ctx[64];
        for (int c = 0; c < 64; ++c) ctx[c] = 0.f;
        for (int j = 0; j < NB_; ++j) {
            float w = sc[j] * inv;
            for (int c = 0; c < 64; ++c) ctx[c] += w * vs[j * 64 + c];
        }
        float aacc = rho_b2[0];
        for (int jj = 0; jj < 64; ++jj) {
            float a = rho_b1[jj];
            for (int c = 0; c < 64; ++c) a += rho_w1[jj * 128 + c] * qs[tid * 64 + c];
            for (int c = 0; c < 64; ++c) a += rho_w1[jj * 128 + 64 + c] * ctx[c];
            aacc += gelu_exact(a) * rho_w2[jj];
        }
        alpha_ws[tid] = 1.f / (1.f + expf(-aacc));
    }
}

// ---------------------------------------------------------------------------
// Kernel 2: build bf16 weights in workspace.
//   W1eff[n][0:256)   = corr_w1 hidden block
//   W1eff[n][256:264) = corr_w1 time block folded through time_w (rank-8)
//   W1eff[n][264:288) = 0 (pad K to multiple of 32)
//   bias1[b][n] = corr_b1[n] + W1_t[n]·time_b + W1_s[n]·static_proj[b]
//   W2bf = bf16(corr_w2)
// ---------------------------------------------------------------------------
__global__ __launch_bounds__(256) void prep_weights_kernel(
    const float* __restrict__ corr_w1, const float* __restrict__ corr_b1,
    const float* __restrict__ corr_w2,
    const float* __restrict__ time_w,  const float* __restrict__ time_b,
    const float* __restrict__ sp_ws,
    __bf16* __restrict__ W1eff, __bf16* __restrict__ W2bf,
    float* __restrict__ bias1)
{
    __shared__ float tb_s;
    const int tid = threadIdx.x;
    const int blk = blockIdx.x;
    if (blk < N1) {
        const int n = blk;
        const float* wrow = corr_w1 + (size_t)n * 768;
        W1eff[(size_t)n * KEFF + tid] = (__bf16)wrow[tid];          // hidden block, tid<256
        if (tid < 8) {
            float a = 0.f;
            for (int d = 0; d < 256; ++d) a += wrow[256 + d] * time_w[d * NT_ + tid];
            W1eff[(size_t)n * KEFF + 256 + tid] = (__bf16)a;
        } else if (tid < 32) {
            W1eff[(size_t)n * KEFF + 264 + (tid - 8)] = (__bf16)0.f;
        } else if (tid == 32) {
            float a = 0.f;
            for (int d = 0; d < 256; ++d) a += wrow[256 + d] * time_b[d];
            tb_s = a;
        }
        __syncthreads();
        if (tid < B_) {
            float a = corr_b1[n] + tb_s;
            const float* sp = sp_ws + tid * 256;
            for (int d = 0; d < 256; ++d) a += wrow[512 + d] * sp[d];
            bias1[tid * N1 + n] = a;
        }
    } else {
        const int n = blk - N1;
        for (int k = tid; k < N1; k += 256)
            W2bf[(size_t)n * N1 + k] = (__bf16)corr_w2[(size_t)n * N1 + k];
    }
}

// ---------------------------------------------------------------------------
// Kernel 3: fused main path. 2048 blocks x 256 threads (8 wave32), 128 tokens
// per block. bf16 WMMA 16x16x32, two-stage GEMM fused through LDS, weight
// K-slices double-buffered via global_load_async_to_lds_b128 (ASYNCcnt),
// then LayerNorm + gated residual.
//
// Dynamic LDS layout (bytes):
//   Xs   [128][288] bf16   @0        73728
//   H1s  [128][512] bf16   @73728   131072
//   Ws   2 x [512][32] bf16 @204800   65536  (double-buffered K-slice staging)
//   bias_s float[512]      @270336    2048
//   b2_s   float[256]      @272384    1024
//   lng_s  float[256]      @273408    1024
//   lnb_s  float[256]      @274432    1024   -> total 275456 (< 320KB WGP LDS)
// ---------------------------------------------------------------------------
__device__ __forceinline__ void issue_w_slice(int tid, __bf16* dstbuf,
                                              const __bf16* W, int kt,
                                              int rows, int rowstride) {
    // copy rows x 32 bf16 (rows*4 x 16B) straight into LDS, no VGPR bounce
    for (int i = tid; i < rows * 4; i += 256) {
        int n = i >> 2, part = i & 3;
        async_copy_b128(dstbuf + n * 32 + part * 8,
                        W + (size_t)n * rowstride + kt * 32 + part * 8);
    }
}

__global__ __launch_bounds__(256) void fused_corr_kernel(
    const float* __restrict__ hidden, const float* __restrict__ tf,
    const __bf16* __restrict__ W1eff, const __bf16* __restrict__ W2bf,
    const float* __restrict__ bias1,  const float* __restrict__ alpha_ws,
    const float* __restrict__ corr_b2,
    const float* __restrict__ ln_g,   const float* __restrict__ ln_b,
    float* __restrict__ out)
{
    extern __shared__ char smem[];
    __bf16* Xs     = (__bf16*)(smem);
    __bf16* H1s    = (__bf16*)(smem + 73728);
    __bf16* Ws     = (__bf16*)(smem + 204800);
    float*  bias_s = (float*)(smem + 270336);
    float*  b2_s   = (float*)(smem + 272384);
    float*  lng_s  = (float*)(smem + 273408);
    float*  lnb_s  = (float*)(smem + 274432);

    const int tid   = threadIdx.x;
    const int m0    = blockIdx.x * MT;
    const int batch = m0 / T_;                 // MT divides T, whole tile shares batch
    const float alpha = alpha_ws[batch];

    // ---- stage 0: load X tile as bf16 into LDS: [hidden(256) | tf(8) | pad(24)]
    for (int i = tid; i < (MT * D_) / 4; i += 256) {      // 8192 float4 loads
        int r = i >> 6, c4 = i & 63;
        float4 hv = *(const float4*)(hidden + (size_t)(m0 + r) * D_ + c4 * 4);
        union { __bf16 h[4]; uint2 u; } pk;
        pk.h[0] = (__bf16)hv.x; pk.h[1] = (__bf16)hv.y;
        pk.h[2] = (__bf16)hv.z; pk.h[3] = (__bf16)hv.w;
        *(uint2*)(Xs + r * KEFF + c4 * 4) = pk.u;
    }
    for (int i = tid; i < MT * 2; i += 256) {             // time features
        int r = i >> 1, hf = i & 1;
        float4 tv = *(const float4*)(tf + (size_t)(m0 + r) * NT_ + hf * 4);
        union { __bf16 h[4]; uint2 u; } pk;
        pk.h[0] = (__bf16)tv.x; pk.h[1] = (__bf16)tv.y;
        pk.h[2] = (__bf16)tv.z; pk.h[3] = (__bf16)tv.w;
        *(uint2*)(Xs + r * KEFF + 256 + hf * 4) = pk.u;
    }
    for (int i = tid; i < MT * 6; i += 256) {             // zero pad cols 264..287
        int r = i / 6, j = i % 6;
        *(uint2*)(Xs + r * KEFF + 264 + j * 4) = make_uint2(0u, 0u);
    }
    for (int i = tid; i < N1; i += 256) bias_s[i] = bias1[batch * N1 + i];
    b2_s[tid]  = corr_b2[tid & 255];
    lng_s[tid] = ln_g[tid & 255];
    lnb_s[tid] = ln_b[tid & 255];

    // pipeline prologue: slice 0 of W1eff -> buffer 0 (async, overlaps nothing yet)
    issue_w_slice(tid, Ws, W1eff, 0, N1, KEFF);

    const int wv   = tid >> 5;       // wave id 0..7 -> owns M rows [16*wv, 16*wv+16)
    const int lane = tid & 31;
    const int hh   = lane >> 4;      // lane half (wave32 WMMA layout)
    const int l16  = lane & 15;
    const int row  = wv * 16 + l16;  // A-fragment row for this lane

    const v8f vzero = {0.f, 0.f, 0.f, 0.f, 0.f, 0.f, 0.f, 0.f};

    // ---- stage 1: H1 = gelu(X @ W1effT + bias1[b])  (K=288, N=512, two N halves)
    // Weight slices form one flat 34-step async pipeline: 2x9 (W1) + 16 (W2).
    for (int p = 0; p < 2; ++p) {
        v8f acc[16];
        #pragma unroll
        for (int nt = 0; nt < 16; ++nt) acc[nt] = vzero;

        for (int kt = 0; kt < KEFF / 32; ++kt) {          // 9 k-steps
            const int s = p * 9 + kt;                     // pipeline step
            wait_async0();                                // my slice-s copies landed
            __syncthreads();                              // all landed; step s-1 compute done
            if (s + 1 < 18)
                issue_w_slice(tid, Ws + ((s + 1) & 1) * WS_HALF, W1eff, (s + 1) % 9, N1, KEFF);
            else                                          // s+1==18: first W2 slice
                issue_w_slice(tid, Ws + ((s + 1) & 1) * WS_HALF, W2bf, 0, N2, N1);
            const __bf16* Wcur = Ws + (s & 1) * WS_HALF;

            V16B A;                                       // 16x32 bf16 A fragment
            const __bf16* ab = Xs + row * KEFF + kt * 32;
            A.q[0] = *(const int4*)(ab + hh * 8);
            A.q[1] = *(const int4*)(ab + 16 + hh * 8);
            #pragma unroll
            for (int nt = 0; nt < 16; ++nt) {
                int n = p * 256 + nt * 16 + l16;
                V16B Bf;                                  // 32x16 bf16 B fragment
                const __bf16* bb = Wcur + n * 32 + hh * 8;
                Bf.q[0] = *(const int4*)(bb);
                Bf.q[1] = *(const int4*)(bb + 16);
                acc[nt] = __builtin_amdgcn_wmma_f32_16x16x32_bf16(
                    false, A.v, false, Bf.v, (short)0, acc[nt], false, false);
            }
        }
        // epilogue: + bias, exact GELU, bf16 -> H1s (overlaps in-flight async fill)
        #pragma unroll
        for (int nt = 0; nt < 16; ++nt) {
            int n = p * 256 + nt * 16 + l16;
            float bs = bias_s[n];
            #pragma unroll
            for (int r = 0; r < 8; ++r) {
                float x = acc[nt][r] + bs;
                H1s[(wv * 16 + hh * 8 + r) * N1 + n] = (__bf16)gelu_exact(x);
            }
        }
    }

    // ---- stage 2: delta = H1 @ W2T + corr_b2   (K=512, N=256)
    v8f acc2[16];
    #pragma unroll
    for (int nt = 0; nt < 16; ++nt) acc2[nt] = vzero;

    for (int kt = 0; kt < N1 / 32; ++kt) {                 // 16 k-steps
        const int s = 18 + kt;
        wait_async0();
        __syncthreads();                                    // H1s complete / other buf free
        if (kt + 1 < N1 / 32)
            issue_w_slice(tid, Ws + ((s + 1) & 1) * WS_HALF, W2bf, kt + 1, N2, N1);
        const __bf16* Wcur = Ws + (s & 1) * WS_HALF;

        V16B A;
        const __bf16* ab = H1s + row * N1 + kt * 32;
        A.q[0] = *(const int4*)(ab + hh * 8);
        A.q[1] = *(const int4*)(ab + 16 + hh * 8);
        #pragma unroll
        for (int nt = 0; nt < 16; ++nt) {
            int n = nt * 16 + l16;
            V16B Bf;
            const __bf16* bb = Wcur + n * 32 + hh * 8;
            Bf.q[0] = *(const int4*)(bb);
            Bf.q[1] = *(const int4*)(bb + 16);
            acc2[nt] = __builtin_amdgcn_wmma_f32_16x16x32_bf16(
                false, A.v, false, Bf.v, (short)0, acc2[nt], false, false);
        }
    }

    // add output bias before LayerNorm statistics
    #pragma unroll
    for (int nt = 0; nt < 16; ++nt) {
        float b2v = b2_s[nt * 16 + l16];
        #pragma unroll
        for (int r = 0; r < 8; ++r) acc2[nt][r] += b2v;
    }

    // ---- LayerNorm + gated residual. Each row's 256 values live across the 16
    // lanes of one half (wave32): butterfly-reduce with xor masks 1,2,4,8.
    #pragma unroll
    for (int r = 0; r < 8; ++r) {
        float s1 = 0.f, s2 = 0.f;
        #pragma unroll
        for (int nt = 0; nt < 16; ++nt) { float x = acc2[nt][r]; s1 += x; s2 += x * x; }
        s1 += __shfl_xor(s1, 1, 32);  s2 += __shfl_xor(s2, 1, 32);
        s1 += __shfl_xor(s1, 2, 32);  s2 += __shfl_xor(s2, 2, 32);
        s1 += __shfl_xor(s1, 4, 32);  s2 += __shfl_xor(s2, 4, 32);
        s1 += __shfl_xor(s1, 8, 32);  s2 += __shfl_xor(s2, 8, 32);
        float mu   = s1 * (1.f / 256.f);
        float var  = s2 * (1.f / 256.f) - mu * mu;
        float rstd = rsqrtf(var + 1e-5f);
        int gm = m0 + wv * 16 + hh * 8 + r;
        #pragma unroll
        for (int nt = 0; nt < 16; ++nt) {
            int n = nt * 16 + l16;
            float y = (acc2[nt][r] - mu) * rstd * lng_s[n] + lnb_s[n];
            out[(size_t)gm * D_ + n] = hidden[(size_t)gm * D_ + n] + alpha * y;
        }
    }
}

// ---------------------------------------------------------------------------
extern "C" void kernel_launch(void* const* d_in, const int* in_sizes, int n_in,
                              void* d_out, int out_size, void* d_ws, size_t ws_size,
                              hipStream_t stream)
{
    const float* hidden  = (const float*)d_in[0];
    const float* tf      = (const float*)d_in[1];
    const float* sf      = (const float*)d_in[2];
    const float* bank    = (const float*)d_in[3];
    const float* phi_w1  = (const float*)d_in[4];
    const float* phi_b1  = (const float*)d_in[5];
    const float* phi_w2  = (const float*)d_in[6];
    const float* phi_b2  = (const float*)d_in[7];
    const float* vproj_w = (const float*)d_in[8];
    const float* vproj_b = (const float*)d_in[9];
    const float* rho_w1  = (const float*)d_in[10];
    const float* rho_b1  = (const float*)d_in[11];
    const float* rho_w2  = (const float*)d_in[12];
    const float* rho_b2  = (const float*)d_in[13];
    const float* time_w  = (const float*)d_in[14];
    const float* time_b  = (const float*)d_in[15];
    const float* stat_w  = (const float*)d_in[16];
    const float* stat_b  = (const float*)d_in[17];
    const float* corr_w1 = (const float*)d_in[18];
    const float* corr_b1 = (const float*)d_in[19];
    const float* corr_w2 = (const float*)d_in[20];
    const float* corr_b2 = (const float*)d_in[21];
    const float* ln_g    = (const float*)d_in[22];
    const float* ln_b    = (const float*)d_in[23];

    // workspace layout (bytes): alpha[64]f32 @0 | static_proj[64x256]f32 @256 |
    // bias1[64x512]f32 @65792 | W1eff[512x288]bf16 @196864 | W2bf[256x512]bf16 @491776
    char* ws = (char*)d_ws;
    float*  alpha_ws = (float*)(ws + 0);
    float*  sp_ws    = (float*)(ws + 256);
    float*  bias1    = (float*)(ws + 65792);
    __bf16* W1eff    = (__bf16*)(ws + 196864);
    __bf16* W2bf     = (__bf16*)(ws + 491776);

    small_path_kernel<<<1, 256, 147456, stream>>>(
        sf, bank, phi_w1, phi_b1, phi_w2, phi_b2, vproj_w, vproj_b,
        rho_w1, rho_b1, rho_w2, rho_b2, stat_w, stat_b, sp_ws, alpha_ws);

    prep_weights_kernel<<<768, 256, 0, stream>>>(
        corr_w1, corr_b1, corr_w2, time_w, time_b, sp_ws, W1eff, W2bf, bias1);

    fused_corr_kernel<<<(B_ * T_) / MT, 256, 275456, stream>>>(
        hidden, tf, W1eff, W2bf, bias1, alpha_ws, corr_b2, ln_g, ln_b, (float*)d_out);
}
